// CopyDecoder_40243843563834
// MI455X (gfx1250) — compile-verified, matrix operand order checked by reading, not run on hardware
//
#include <hip/hip_runtime.h>
#include <hip/hip_bf16.h>
#include <math.h>

// ---------------- CDNA5 WMMA types ----------------
typedef __attribute__((ext_vector_type(16))) __bf16 v16bf;
typedef __attribute__((ext_vector_type(8)))  __bf16 v8bf;
typedef __attribute__((ext_vector_type(8)))  float  v8f;

#define VOCAB  50000
#define EMBED  256
#define HIDDEN 512
#define BATCH  128
#define SEQ    400
#define GATES  (3 * HIDDEN)     // 1536
#define XDIM   (EMBED + HIDDEN) // 768

// CDNA5 16-bit A/B fragment layout per 32-wide K step:
//   lanes 0-15 : VGPR0-3 hold K=0..7 (pairs), VGPR4-7 hold K=16..23
//   lanes 16-31: VGPR0-3 hold K=8..15,        VGPR4-7 hold K=24..31
// => element e (0..15) of the v16bf in lane L maps to
//    k = k0 + (L/16)*8 + (e<8 ? e : 16 + (e-8))

// fp32 source: 4x float4 loads + pack-convert to bf16
__device__ __forceinline__ v16bf load_frag_f32(const float* __restrict__ row,
                                               int k0, int halfsel) {
    const float4* p = (const float4*)(row + k0 + halfsel * 8);
    float4 q0 = p[0], q1 = p[1];   // + 0..7
    float4 q2 = p[4], q3 = p[5];   // + 16..23
    v16bf v;
    v[0]  = (__bf16)q0.x; v[1]  = (__bf16)q0.y; v[2]  = (__bf16)q0.z; v[3]  = (__bf16)q0.w;
    v[4]  = (__bf16)q1.x; v[5]  = (__bf16)q1.y; v[6]  = (__bf16)q1.z; v[7]  = (__bf16)q1.w;
    v[8]  = (__bf16)q2.x; v[9]  = (__bf16)q2.y; v[10] = (__bf16)q2.z; v[11] = (__bf16)q2.w;
    v[12] = (__bf16)q3.x; v[13] = (__bf16)q3.y; v[14] = (__bf16)q3.z; v[15] = (__bf16)q3.w;
    return v;
}

// bf16 source: two raw b128 loads, no conversion VALU
__device__ __forceinline__ v16bf load_frag_bf16(const __bf16* __restrict__ row,
                                                int k0, int halfsel) {
    const __bf16* p = row + k0 + halfsel * 8;
    v8bf lo = *(const v8bf*)(p);       // + 0..7
    v8bf hi = *(const v8bf*)(p + 16);  // + 16..23
    return __builtin_shufflevector(lo, hi, 0, 1, 2, 3, 4, 5, 6, 7,
                                           8, 9, 10, 11, 12, 13, 14, 15);
}

// ---------------- K0: fp32 -> bf16 pre-convert (used for Wc) ----------------
__global__ void cvt_bf16_kernel(const float* __restrict__ in,
                                __bf16* __restrict__ out, int n) {
    int i = blockIdx.x * blockDim.x + threadIdx.x;
    if (i < n) out[i] = (__bf16)in[i];
}

// ---------------- K1: x = [embed[idx], order ? weighted : 0] ----------------
__global__ void build_x_kernel(const int* __restrict__ input_idx,
                               const float* __restrict__ embed_W,
                               const float* __restrict__ weighted,
                               const int* __restrict__ order,
                               float* __restrict__ x) {
    int i = blockIdx.x * blockDim.x + threadIdx.x;
    if (i >= BATCH * XDIM) return;
    int b = i / XDIM, e = i % XDIM;
    float v;
    if (e < EMBED) v = embed_W[(size_t)input_idx[b] * EMBED + e];
    else           v = (order[0] != 0) ? weighted[b * HIDDEN + (e - EMBED)] : 0.0f;
    x[i] = v;
}

// ---------------- Generic WMMA GEMM: C[M,N] = A[M,K] @ W[N,K]^T + bias ------
// One wave per 16x16 tile. K, LDC compile-time: 4x-unrolled K loop, immediate-offset
// stores. __launch_bounds__(32,1) lifts the default VGPR cap so the unrolled
// load clusters never spill.
template <int K, int LDC>
__global__ void __launch_bounds__(32, 1)
wmma_gemm_bf16(const float* __restrict__ A,
               const float* __restrict__ W,
               const float* __restrict__ bias,
               float* __restrict__ C) {
    constexpr int KSTEPS = K / 32;
    const int lane = threadIdx.x;          // 0..31 (wave32)
    const int nb   = blockIdx.x * 16;
    const int mb   = blockIdx.y * 16;
    const int half = lane >> 4, l16 = lane & 15;

    const float* arow = A + (size_t)(mb + l16) * K;
    const float* wrow = W + (size_t)(nb + l16) * K;

    v8f acc = {};
    #pragma unroll 4
    for (int ks = 0; ks < KSTEPS; ++ks) {
        if (ks * 32 + 96 < K) __builtin_prefetch(wrow + ks * 32 + 96, 0, 0);
        v16bf af = load_frag_f32(arow, ks * 32, half);
        v16bf bf = load_frag_f32(wrow, ks * 32, half);
        acc = __builtin_amdgcn_wmma_f32_16x16x32_bf16(
                  false, af, false, bf, (short)0, acc, false, false);
    }

    const float bv = bias[nb + l16];
    float* crow = C + (size_t)(mb + half * 8) * LDC + (nb + l16);
    #pragma unroll
    for (int i = 0; i < 8; ++i)              // immediate-offset stores (LDC constexpr)
        crow[(size_t)i * LDC] = acc[i] + bv;
}

// ---------------- K2c: GRU gates -> state ----------------
__device__ __forceinline__ float sigmoidf_(float v) { return 1.0f / (1.0f + __expf(-v)); }

__global__ void gru_gates_kernel(const float* __restrict__ gi,
                                 const float* __restrict__ gh,
                                 const float* __restrict__ h,
                                 float* __restrict__ state) {
    int i = blockIdx.x * blockDim.x + threadIdx.x;
    if (i >= BATCH * HIDDEN) return;
    int b = i / HIDDEN, j = i % HIDDEN;
    const float* gib = gi + (size_t)b * GATES;
    const float* ghb = gh + (size_t)b * GATES;
    float r = sigmoidf_(gib[j]          + ghb[j]);
    float z = sigmoidf_(gib[j + HIDDEN] + ghb[j + HIDDEN]);
    float n = tanhf(gib[j + 2 * HIDDEN] + r * ghb[j + 2 * HIDDEN]);
    state[i] = (1.0f - z) * n + z * h[i];
}

// ---------------- K3: fused score_c ----------------
// One wave per 16-row strip of encoded. The whole 16x512 strip lives in
// registers as 16 bf16 fragments (128 VGPRs; cap lifted via
// __launch_bounds__(32,1) so nothing spills). The 32 h-tiles then run a pure
// global_load_b128 + WMMA loop; Wc is pre-converted to bf16 (no cvt VALU).
__global__ void __launch_bounds__(32, 1)
score_c_kernel(const float* __restrict__ encoded,
               const __bf16* __restrict__ Wc_bf,
               const float* __restrict__ Wc_b,
               const float* __restrict__ state,
               const int* __restrict__ enc_idx,
               float* __restrict__ score_c) {
    constexpr int KSTEPS = HIDDEN / 32;     // 16
    const int lane = threadIdx.x;
    const int tile = blockIdx.x;            // SEQ%16==0 so strips never straddle b
    const int b    = tile / (SEQ / 16);
    const int s0   = (tile % (SEQ / 16)) * 16;
    const int half = lane >> 4, l16 = lane & 15;

    const float* arow = encoded + ((size_t)b * SEQ + s0 + l16) * HIDDEN;

    v16bf afrag[KSTEPS];                    // 128 VGPRs: strip loaded/converted once
    #pragma unroll
    for (int ks = 0; ks < KSTEPS; ++ks)
        afrag[ks] = load_frag_f32(arow, ks * 32, half);

    float partial[8];
    #pragma unroll
    for (int i = 0; i < 8; ++i) partial[i] = 0.0f;

    for (int h0 = 0; h0 < HIDDEN; h0 += 16) {
        const __bf16* wrow = Wc_bf + (size_t)(h0 + l16) * HIDDEN;
        v8f acc = {};
        #pragma unroll
        for (int ks = 0; ks < KSTEPS; ++ks) {
            v16bf bf = load_frag_bf16(wrow, ks * 32, half);
            acc = __builtin_amdgcn_wmma_f32_16x16x32_bf16(
                      false, afrag[ks], false, bf, (short)0, acc, false, false);
        }
        const int   hh = h0 + l16;          // this lane's N column == hidden index
        const float cb = Wc_b[hh];
        const float sb = state[b * HIDDEN + hh];
        #pragma unroll
        for (int i = 0; i < 8; ++i)
            partial[i] += tanhf(acc[i] + cb) * sb;
    }

    // butterfly reduce over the 16 lanes of each half (N dimension)
    #pragma unroll
    for (int off = 8; off >= 1; off >>= 1) {
        #pragma unroll
        for (int i = 0; i < 8; ++i)
            partial[i] += __shfl_xor(partial[i], off, 32);
    }

    if (l16 == 0) {
        #pragma unroll
        for (int i = 0; i < 8; ++i) {
            int s = s0 + i + half * 8;
            int r = b * SEQ + s;
            float v = tanhf(partial[i]);
            if (enc_idx[r] == 0) v += -1000.0f;
            score_c[r] = v;
        }
    }
}

// ---------------- K5: per-row max + sum(exp) over [score_g | score_c] -------
__global__ void row_max_sum_kernel(const float* __restrict__ score_g,
                                   const float* __restrict__ score_c,
                                   float* __restrict__ mz) {
    const int b = blockIdx.x, t = threadIdx.x;   // 256 threads
    __shared__ float red[256];

    float m = -3.4e38f;
    for (int i = t; i < VOCAB; i += 256) m = fmaxf(m, score_g[(size_t)b * VOCAB + i]);
    for (int i = t; i < SEQ;   i += 256) m = fmaxf(m, score_c[b * SEQ + i]);
    red[t] = m; __syncthreads();
    for (int s = 128; s > 0; s >>= 1) {
        if (t < s) red[t] = fmaxf(red[t], red[t + s]);
        __syncthreads();
    }
    m = red[0]; __syncthreads();

    float z = 0.0f;
    for (int i = t; i < VOCAB; i += 256) z += __expf(score_g[(size_t)b * VOCAB + i] - m);
    for (int i = t; i < SEQ;   i += 256) z += __expf(score_c[b * SEQ + i] - m);
    red[t] = z; __syncthreads();
    for (int s = 128; s > 0; s >>= 1) {
        if (t < s) red[t] += red[t + s];
        __syncthreads();
    }
    if (t == 0) { mz[2 * b] = m; mz[2 * b + 1] = red[0]; }
}

// ---------------- K6: in-place prob_g = exp(score_g - m)/Z ----------------
__global__ void finalize_pg_kernel(float* __restrict__ out, const float* __restrict__ mz) {
    size_t i = (size_t)blockIdx.x * blockDim.x + threadIdx.x;
    if (i >= (size_t)BATCH * VOCAB) return;
    int b = (int)(i / VOCAB);
    out[i] = __expf(out[i] - mz[2 * b]) / mz[2 * b + 1];
}

// ---------------- K7: scatter prob_c into out + copy-attn weighted_out ------
__global__ void copy_attn_kernel(const float* __restrict__ score_c,
                                 const int* __restrict__ enc_idx,
                                 const int* __restrict__ input_idx,
                                 const float* __restrict__ mz,
                                 const float* __restrict__ encoded,
                                 float* __restrict__ out,
                                 float* __restrict__ weighted_out) {
    const int b = blockIdx.x, t = threadIdx.x;   // 512 threads
    __shared__ float attn[SEQ];
    __shared__ int   cnt;
    if (t == 0) cnt = 0;
    __syncthreads();

    const int   ii   = input_idx[b];
    const float m    = mz[2 * b];
    const float invZ = 1.0f / mz[2 * b + 1];

    if (t < SEQ && enc_idx[b * SEQ + t] == ii) atomicAdd(&cnt, 1);
    __syncthreads();
    const float coef = (cnt > 1) ? 1.0f / (float)cnt : 1.0f;

    if (t < SEQ) {
        const int   idx = enc_idx[b * SEQ + t];
        const float p   = __expf(score_c[b * SEQ + t] - m) * invZ;
        atomicAdd(&out[(size_t)b * VOCAB + idx], p);
        attn[t] = (idx == ii) ? p * coef : 0.0f;
    }
    __syncthreads();

    // weighted_out[b, t] = sum_s attn[s] * encoded[b, s, t]  (attn is sparse)
    float acc = 0.0f;
    for (int s = 0; s < SEQ; ++s) {
        float a = attn[s];
        if (a != 0.0f) acc += a * encoded[((size_t)b * SEQ + s) * HIDDEN + t];
    }
    weighted_out[b * HIDDEN + t] = acc;
}

// ---------------- host launcher ----------------
extern "C" void kernel_launch(void* const* d_in, const int* in_sizes, int n_in,
                              void* d_out, int out_size, void* d_ws, size_t ws_size,
                              hipStream_t stream) {
    const int*   input_idx  = (const int*)  d_in[0];
    const float* encoded    = (const float*)d_in[1];
    const int*   enc_idx    = (const int*)  d_in[2];
    const float* prev_state = (const float*)d_in[3];
    const float* weighted   = (const float*)d_in[4];
    const int*   order      = (const int*)  d_in[5];
    const float* embed_W    = (const float*)d_in[6];
    const float* W_ih       = (const float*)d_in[7];
    const float* W_hh       = (const float*)d_in[8];
    const float* b_ih       = (const float*)d_in[9];
    const float* b_hh       = (const float*)d_in[10];
    const float* Wo_W       = (const float*)d_in[11];
    const float* Wo_b       = (const float*)d_in[12];
    const float* Wc_W       = (const float*)d_in[13];
    const float* Wc_b       = (const float*)d_in[14];

    float* out   = (float*)d_out;                       // (B, VOCAB)
    float* state = out   + (size_t)BATCH * VOCAB;       // (B, HIDDEN)
    float* wout  = state + (size_t)BATCH * HIDDEN;      // (B, HIDDEN)

    float* ws = (float*)d_ws;
    float*  x       = ws;                               // B * XDIM
    float*  gi      = x  + (size_t)BATCH * XDIM;        // B * GATES
    float*  gh      = gi + (size_t)BATCH * GATES;       // B * GATES
    float*  score_c = gh + (size_t)BATCH * GATES;       // B * SEQ
    float*  mz      = score_c + (size_t)BATCH * SEQ;    // 2 * B
    __bf16* Wc_bf   = (__bf16*)(mz + 2 * BATCH);        // HIDDEN * HIDDEN bf16 (16B-aligned)

    // 0) pre-convert Wc to bf16 (512 KB, reused by all 3200 score_c waves)
    cvt_bf16_kernel<<<(HIDDEN * HIDDEN + 255) / 256, 256, 0, stream>>>(
        Wc_W, Wc_bf, HIDDEN * HIDDEN);

    // 1) x = [embed[idx], order ? weighted : 0]
    build_x_kernel<<<(BATCH * XDIM + 255) / 256, 256, 0, stream>>>(
        input_idx, embed_W, weighted, order, x);

    // 2) gi = x @ W_ih^T + b_ih ; gh = h @ W_hh^T + b_hh   (WMMA bf16)
    wmma_gemm_bf16<XDIM, GATES><<<dim3(GATES / 16, BATCH / 16), 32, 0, stream>>>(
        x, W_ih, b_ih, gi);
    wmma_gemm_bf16<HIDDEN, GATES><<<dim3(GATES / 16, BATCH / 16), 32, 0, stream>>>(
        prev_state, W_hh, b_hh, gh);

    // 3) GRU gates -> state (written directly into d_out)
    gru_gates_kernel<<<(BATCH * HIDDEN + 255) / 256, 256, 0, stream>>>(
        gi, gh, prev_state, state);

    // 4) fused score_c (WMMA bf16; strip register-resident, Wc pre-converted)
    score_c_kernel<<<BATCH * (SEQ / 16), 32, 0, stream>>>(
        encoded, Wc_bf, Wc_b, state, enc_idx, score_c);

    // 5) score_g = state @ Wo^T + Wo_b, straight into d_out prob region
    wmma_gemm_bf16<HIDDEN, VOCAB><<<dim3(VOCAB / 16, BATCH / 16), 32, 0, stream>>>(
        state, Wo_W, Wo_b, out);

    // 6) softmax statistics over [score_g | score_c]
    row_max_sum_kernel<<<BATCH, 256, 0, stream>>>(out, score_c, mz);

    // 7) prob_g in place
    finalize_pg_kernel<<<(int)(((size_t)BATCH * VOCAB + 255) / 256), 256, 0, stream>>>(
        out, mz);

    // 8) scatter prob_c into out, copy-attention, weighted_out
    copy_attn_kernel<<<BATCH, 512, 0, stream>>>(
        score_c, enc_idx, input_idx, mz, encoded, out, wout);
}
